// GNNCombiner_89781996356248
// MI455X (gfx1250) — matrix operand order repaired and checked by reading, not compiled, not run on hardware
//
#include <hip/hip_runtime.h>
#include <cstddef>

// ---------------------------------------------------------------------------
// CDNA5 (gfx1250, wave32) bf16 WMMA implementation of the GNN+Transformer net.
// All GEMMs use v_wmma_f32_16x16x32_bf16. Activations bf16, accumulate fp32.
// Weights are pre-swizzled once per launch into WMMA B-fragment-major layout
// (32 lanes x 16 contiguous bf16 = 1KB/fragment) so the GEMM hot loop does
// only coalesced global_load_b128 (L2-resident) + WMMA: no LDS, no barriers.
// ---------------------------------------------------------------------------

typedef __attribute__((ext_vector_type(16))) __bf16 bf16x16;
typedef __attribute__((ext_vector_type(8)))  __bf16 bf16x8;
typedef __attribute__((ext_vector_type(8)))  float  f32x8;

#define NROWS   (512 * 64 * 16)   // B*L*K = 524288
#define UROWS   (512 * 16)        // B*K   = 8192
#define EPSLN   1e-5f

// ------------------------------ utility ------------------------------------

__global__ void zero_f32_kernel(float* p, int n) {
    int i = blockIdx.x * blockDim.x + threadIdx.x;
    if (i < n) p[i] = 0.0f;
}

// Pack fp32 row-major W[K,M] into bf16 WMMA B-fragment-major layout:
// frag = (k/32)*(M/16) + (c/16); within frag, lane = (k-half<<4)|col holds
// 16 contiguous bf16 covering K = kb..kb+15 of column c.
__global__ void pack_w_kernel(const float* __restrict__ W, __bf16* __restrict__ Wp,
                              int K, int M) {
    int t = blockIdx.x * blockDim.x + threadIdx.x;
    int total = (K >> 5) * (M >> 4) * 32;
    if (t >= total) return;
    int lane = t & 31;
    int frag = t >> 5;
    int mtiles = M >> 4;
    int colTile = frag % mtiles;
    int chunk   = frag / mtiles;
    int col = colTile * 16 + (lane & 15);
    int kb  = chunk * 32 + ((lane >> 4) << 4);
    __bf16* dst = Wp + (size_t)frag * 512 + lane * 16;
#pragma unroll
    for (int e = 0; e < 16; ++e) dst[e] = (__bf16)W[(size_t)(kb + e) * M + col];
}

// ------------------------- input MLP layer 1 (5 -> 128, relu) --------------

__global__ __launch_bounds__(256)
void input_mlp1_kernel(const float* __restrict__ s_hat, const float* __restrict__ cqi,
                       const float* __restrict__ bits,  const float* __restrict__ snr,
                       const float* __restrict__ w1,    const float* __restrict__ b1,
                       __bf16* __restrict__ H1, int N) {
    __shared__ float sw[5 * 128];
    __shared__ float sb[128];
    for (int i = threadIdx.x; i < 640; i += 256) sw[i] = w1[i];
    if (threadIdx.x < 128) sb[threadIdx.x] = b1[threadIdx.x];
    __syncthreads();
    int r = blockIdx.x * 256 + threadIdx.x;
    if (r >= N) return;
    float x0 = s_hat[r * 2], x1 = s_hat[r * 2 + 1];
    float x2 = cqi[r], x3 = bits[r], x4 = snr[r];
    __bf16* o = H1 + (size_t)r * 128;
#pragma unroll 4
    for (int d = 0; d < 128; ++d) {
        float a = sb[d] + x0 * sw[d] + x1 * sw[128 + d] + x2 * sw[256 + d]
                + x3 * sw[384 + d] + x4 * sw[512 + d];
        o[d] = (__bf16)fmaxf(a, 0.0f);
    }
}

// ------------------------- generic bf16 WMMA GEMM ---------------------------
// C[N,M] = act( A[N,K] @ W[K,M] + bias )  (+ resid)
// Block = 128 threads (4 waves); wave = 32 rows x 64 cols (2x4 wmma accs).
// Wp is fragment-major pre-swizzled weights (see pack_w_kernel).
// CONCAT: for k>=128 read A2[bkrow(row), k-128] (mean-message broadcast).

template <bool RELU, bool RESID, bool CONCAT>
__global__ __launch_bounds__(128)
void gemm_bf16_kernel(const __bf16* __restrict__ A,
                      const __bf16* __restrict__ A2,
                      const __bf16* __restrict__ Wp,
                      const float*  __restrict__ bias,
                      const __bf16* __restrict__ resid,
                      __bf16* __restrict__ out,
                      int N, int Kdim, int M, int LK, int KU) {
    const int lane = threadIdx.x & 31;
    const int wave = threadIdx.x >> 5;
    const int rowBase = blockIdx.x * 128 + wave * 32;
    const int colBlk  = blockIdx.y * 64;
    const int ldA = CONCAT ? 128 : Kdim;
    const int mtiles = M >> 4;

    const int aRow0 = rowBase + (lane & 15);      // A frag rows (lane halves share)
    const int aRow1 = aRow0 + 16;
    int a2Row0 = 0, a2Row1 = 0;
    if (CONCAT) {
        a2Row0 = (aRow0 / LK) * KU + (aRow0 % KU);
        a2Row1 = (aRow1 / LK) * KU + (aRow1 % KU);
    }
    const int kHalf = ((lane >> 4) & 1) * 16;     // A-frag K: contiguous 16 per lane half

    f32x8 acc[2][4] = {};

    for (int kc = 0; kc < Kdim; kc += 32) {
        // --- A fragments: two contiguous 8-element chunks per lane, 2 rows ---
        bf16x16 aF0, aF1;
        {
            const __bf16 *ap0, *ap1;
            int kl;
            if (CONCAT && kc >= 128) {
                ap0 = A2 + (size_t)a2Row0 * 128;
                ap1 = A2 + (size_t)a2Row1 * 128;
                kl = kc - 128;
            } else {
                ap0 = A + (size_t)aRow0 * ldA;
                ap1 = A + (size_t)aRow1 * ldA;
                kl = kc;
            }
            bf16x8 lo0 = *(const bf16x8*)(ap0 + kl + kHalf);
            bf16x8 hi0 = *(const bf16x8*)(ap0 + kl + kHalf + 8);
            bf16x8 lo1 = *(const bf16x8*)(ap1 + kl + kHalf);
            bf16x8 hi1 = *(const bf16x8*)(ap1 + kl + kHalf + 8);
#pragma unroll
            for (int i = 0; i < 8; ++i) {
                aF0[i] = lo0[i]; aF0[8 + i] = hi0[i];
                aF1[i] = lo1[i]; aF1[8 + i] = hi1[i];
            }
            if (kc + 32 < Kdim) {
                __builtin_prefetch(ap0 + kl + 32, 0, 1);
                __builtin_prefetch(ap1 + kl + 32, 0, 1);
            }
        }

        // --- B fragments: one coalesced 32B/lane load each (L2-resident) ---
        const __bf16* wb = Wp + ((size_t)(kc >> 5) * mtiles + (colBlk >> 4)) * 512
                              + lane * 16;
#pragma unroll
        for (int ct = 0; ct < 4; ++ct) {
            bf16x16 bF = *(const bf16x16*)(wb + (size_t)ct * 512);
            acc[0][ct] = __builtin_amdgcn_wmma_f32_16x16x32_bf16(
                false, aF0, false, bF, (short)0, acc[0][ct], false, false);
            acc[1][ct] = __builtin_amdgcn_wmma_f32_16x16x32_bf16(
                false, aF1, false, bF, (short)0, acc[1][ct], false, false);
        }
    }

    // epilogue (C layout: lane-half picks M offset 0/8, VGPR index = M row)
    const int on = lane & 15;
    const int om = (lane >> 4) * 8;
#pragma unroll
    for (int rb = 0; rb < 2; ++rb) {
#pragma unroll
        for (int ct = 0; ct < 4; ++ct) {
#pragma unroll
            for (int v = 0; v < 8; ++v) {
                int r = rowBase + rb * 16 + om + v;
                int c = colBlk + ct * 16 + on;
                float val = acc[rb][ct][v] + bias[c];
                if (RELU)  val = fmaxf(val, 0.0f);
                if (RESID) val += (float)resid[(size_t)r * M + c];
                out[(size_t)r * M + c] = (__bf16)val;
            }
        }
    }
}

// ------------------------- BatchNorm (training batch stats) -----------------

__global__ __launch_bounds__(256)
void bn_stats_kernel(const __bf16* __restrict__ H, float* __restrict__ stats, int N) {
    const int c   = threadIdx.x & 127;
    const int sub = threadIdx.x >> 7;
    int row = blockIdx.x * 2 + sub;
    const int stride = gridDim.x * 2;
    float s = 0.0f, s2 = 0.0f;
    for (; row < N; row += stride) {
        float v = (float)H[(size_t)row * 128 + c];
        s += v; s2 += v * v;
    }
    atomicAdd(&stats[c], s);
    atomicAdd(&stats[128 + c], s2);
}

__global__ __launch_bounds__(256)
void bn_apply_kernel(__bf16* __restrict__ H, const float* __restrict__ stats,
                     const float* __restrict__ g, const float* __restrict__ b,
                     long total, float invN) {
    long i = (long)blockIdx.x * blockDim.x + threadIdx.x;
    if (i >= total) return;
    int c = (int)(i & 127);
    float mean = stats[c] * invN;
    float var  = stats[128 + c] * invN - mean * mean;
    float sc   = g[c] * rsqrtf(var + EPSLN);
    float sh   = b[c] - mean * sc;
    H[i] = (__bf16)((float)H[i] * sc + sh);
}

// ------------------------- mean over L (GNN message mean) -------------------

__global__ __launch_bounds__(256)
void mean_l_kernel(const __bf16* __restrict__ MSG, __bf16* __restrict__ MM) {
    int i = blockIdx.x * blockDim.x + threadIdx.x;   // over B*K*128
    if (i >= 512 * 16 * 128) return;
    int d = i & 127;
    int k = (i >> 7) & 15;
    int b = i >> 11;
    float s = 0.0f;
    const __bf16* base = MSG + ((size_t)(b * 64) * 16 + k) * 128 + d;
#pragma unroll 4
    for (int l = 0; l < 64; ++l) s += (float)base[(size_t)l * 16 * 128];
    MM[((size_t)b * 16 + k) * 128 + d] = (__bf16)(s * (1.0f / 64.0f));
}

// ------------------------- row LayerNorm (wave per row, D=128) --------------

template <bool RESID>
__global__ __launch_bounds__(256)
void ln_rows_kernel(const __bf16* __restrict__ X, const __bf16* __restrict__ R2,
                    const float* __restrict__ g, const float* __restrict__ bta,
                    __bf16* __restrict__ out, int Rows) {
    const int lane = threadIdx.x & 31;
    const int wave = threadIdx.x >> 5;
    const int row  = blockIdx.x * 8 + wave;
    if (row >= Rows) return;
    const __bf16* xp = X + (size_t)row * 128;
    float v[4];
#pragma unroll
    for (int i = 0; i < 4; ++i) {
        int c = lane * 4 + i;
        float t = (float)xp[c];
        if (RESID) t += (float)R2[(size_t)row * 128 + c];
        v[i] = t;
    }
    float s  = v[0] + v[1] + v[2] + v[3];
    float s2 = v[0]*v[0] + v[1]*v[1] + v[2]*v[2] + v[3]*v[3];
#pragma unroll
    for (int off = 16; off >= 1; off >>= 1) {
        s  += __shfl_xor(s,  off, 32);
        s2 += __shfl_xor(s2, off, 32);
    }
    float mean = s * (1.0f / 128.0f);
    float var  = s2 * (1.0f / 128.0f) - mean * mean;
    float rstd = rsqrtf(var + EPSLN);
#pragma unroll
    for (int i = 0; i < 4; ++i) {
        int c = lane * 4 + i;
        out[(size_t)row * 128 + c] = (__bf16)((v[i] - mean) * rstd * g[c] + bta[c]);
    }
}

// ------------------------- attention pooling over L -------------------------

__global__ __launch_bounds__(64)
void pool_kernel(const __bf16* __restrict__ H, const __bf16* __restrict__ AP1,
                 const float* __restrict__ aw2, const float* __restrict__ ab2,
                 __bf16* __restrict__ U) {
    __shared__ float wgt[64];
    const int b = blockIdx.x >> 4;
    const int k = blockIdx.x & 15;
    const int l = threadIdx.x;
    const size_t row = (size_t)(b * 64 + l) * 16 + k;
    float acc = ab2[0];
    const __bf16* ap = AP1 + row * 64;
#pragma unroll 8
    for (int d = 0; d < 64; ++d) acc += (float)ap[d] * aw2[d];
    wgt[l] = acc;
    __syncthreads();
    float mx = -1e30f;
    for (int j = 0; j < 64; ++j) mx = fmaxf(mx, wgt[j]);
    float e = __expf(acc - mx);
    __syncthreads();
    wgt[l] = e;
    __syncthreads();
    float sm = 0.0f;
    for (int j = 0; j < 64; ++j) sm += wgt[j];
    __syncthreads();
    wgt[l] = e / sm;
    __syncthreads();
    for (int d = threadIdx.x; d < 128; d += 64) {
        float s = 0.0f;
        for (int j = 0; j < 64; ++j)
            s += wgt[j] * (float)H[((size_t)(b * 64 + j) * 16 + k) * 128 + d];
        U[((size_t)b * 16 + k) * 128 + d] = (__bf16)s;
    }
}

// ------------------------- 16-token multi-head attention --------------------
// One wave per (batch, head). seq=16, DH=32 -> q@k^T is ONE wmma 16x16x32,
// att@v is two wmma with K zero-padded 16->32.

__global__ __launch_bounds__(32)
void attn16_kernel(const __bf16* __restrict__ qkv,   // (B*16) x 384
                   __bf16* __restrict__ attO) {      // (B*16) x 128
    __shared__ __bf16 sAtt[16][16];
    const int lane = threadIdx.x;
    const int b = blockIdx.x >> 2;
    const int h = blockIdx.x & 3;
    const int row0 = b * 16;
    const float scale = 0.17677669529663687f;        // 1/sqrt(32)

    const int m     = lane & 15;
    const int kHalf = ((lane >> 4) & 1) * 16;

    // Q fragment (A 16x32)
    bf16x16 qF;
    {
        const __bf16* qp = qkv + (size_t)(row0 + m) * 384 + h * 32;
        bf16x8 lo = *(const bf16x8*)(qp + kHalf);
        bf16x8 hi = *(const bf16x8*)(qp + kHalf + 8);
#pragma unroll
        for (int i = 0; i < 8; ++i) { qF[i] = lo[i]; qF[8 + i] = hi[i]; }
    }
    // K^T fragment (B 32x16): lane = key position, elements = head dims
    bf16x16 kF;
    {
        const int col = lane & 15;
        const int kb  = (lane >> 4) << 4;
        const __bf16* kp = qkv + (size_t)(row0 + col) * 384 + 128 + h * 32 + kb;
        bf16x8 lo = *(const bf16x8*)(kp);
        bf16x8 hi = *(const bf16x8*)(kp + 8);
#pragma unroll
        for (int i = 0; i < 8; ++i) { kF[i] = lo[i]; kF[8 + i] = hi[i]; }
    }

    f32x8 s = {};
    s = __builtin_amdgcn_wmma_f32_16x16x32_bf16(false, qF, false, kF,
                                                (short)0, s, false, false);

    // softmax over keys (lanes 0..15 of each half hold one score row)
#pragma unroll
    for (int v = 0; v < 8; ++v) {
        float x = s[v] * scale;
        float mx = x;
#pragma unroll
        for (int off = 8; off >= 1; off >>= 1) mx = fmaxf(mx, __shfl_xor(mx, off, 32));
        float ex = __expf(x - mx);
        float sm = ex;
#pragma unroll
        for (int off = 8; off >= 1; off >>= 1) sm += __shfl_xor(sm, off, 32);
        s[v] = ex / sm;
    }

    // round-trip through LDS to re-swizzle probs into an A fragment
    const int mo = (lane >> 4) * 8;
    const int n  = lane & 15;
#pragma unroll
    for (int v = 0; v < 8; ++v) sAtt[mo + v][n] = (__bf16)s[v];
    __syncthreads();

    bf16x16 aF;
#pragma unroll
    for (int e = 0; e < 16; ++e)   // lanes <16: K = 0..15 valid; lanes >=16: K = 16..31 pad
        aF[e] = (lane < 16) ? sAtt[m][e] : (__bf16)0.0f;

#pragma unroll
    for (int dt = 0; dt < 2; ++dt) {
        bf16x16 vF;
        if (lane < 16) {   // lanes >=16 correspond to padded K rows -> zero
#pragma unroll
            for (int e = 0; e < 16; ++e)
                vF[e] = qkv[(size_t)(row0 + e) * 384 + 256 + h * 32 + dt * 16 + n];
        } else {
#pragma unroll
            for (int e = 0; e < 16; ++e) vF[e] = (__bf16)0.0f;
        }
        f32x8 o = {};
        o = __builtin_amdgcn_wmma_f32_16x16x32_bf16(false, aF, false, vF,
                                                    (short)0, o, false, false);
#pragma unroll
        for (int v = 0; v < 8; ++v)
            attO[(size_t)(row0 + mo + v) * 128 + h * 32 + dt * 16 + n] = (__bf16)o[v];
    }
}

// ------------------------- output head (64 -> 2) ----------------------------

__global__ __launch_bounds__(256)
void out_head_kernel(const __bf16* __restrict__ O1, const float* __restrict__ ow2,
                     const float* __restrict__ ob2, float* __restrict__ out, int Rows) {
    int r = blockIdx.x * blockDim.x + threadIdx.x;
    if (r >= Rows) return;
    float a0 = ob2[0], a1 = ob2[1];
    const __bf16* p = O1 + (size_t)r * 64;
#pragma unroll 8
    for (int d = 0; d < 64; ++d) {
        float v = (float)p[d];
        a0 += v * ow2[d * 2];
        a1 += v * ow2[d * 2 + 1];
    }
    out[r * 2]     = a0;
    out[r * 2 + 1] = a1;
}

// ---------------------------------------------------------------------------
//                              host side
// ---------------------------------------------------------------------------

static void launch_gemm(hipStream_t st, const __bf16* A, const __bf16* A2,
                        const __bf16* W, const float* bias, const __bf16* resid,
                        __bf16* out, int N, int K, int M, int mode) {
    dim3 g(N / 128, M / 64), b(128);
    switch (mode) {
    case 0: hipLaunchKernelGGL((gemm_bf16_kernel<false, false, false>), g, b, 0, st,
                               A, A2, W, bias, resid, out, N, K, M, 1024, 16); break;
    case 1: hipLaunchKernelGGL((gemm_bf16_kernel<true, false, false>), g, b, 0, st,
                               A, A2, W, bias, resid, out, N, K, M, 1024, 16); break;
    case 2: hipLaunchKernelGGL((gemm_bf16_kernel<true, false, true>), g, b, 0, st,
                               A, A2, W, bias, resid, out, N, K, M, 1024, 16); break;
    case 3: hipLaunchKernelGGL((gemm_bf16_kernel<false, true, false>), g, b, 0, st,
                               A, A2, W, bias, resid, out, N, K, M, 1024, 16); break;
    }
}

static __bf16* pack_weight(hipStream_t st, const float* src, int K, int M,
                           char* base, size_t& off) {
    off = (off + 255) & ~(size_t)255;
    __bf16* dst = (__bf16*)(base + off);
    int elems = K * M;
    off += (size_t)elems * sizeof(__bf16);
    pack_w_kernel<<<(elems / 16 + 255) / 256, 256, 0, st>>>(src, dst, K, M);
    return dst;
}

extern "C" void kernel_launch(void* const* d_in, const int* in_sizes, int n_in,
                              void* d_out, int out_size, void* d_ws, size_t ws_size,
                              hipStream_t stream) {
    (void)in_sizes; (void)n_in; (void)out_size; (void)ws_size;
    const float* P[72];
    for (int i = 0; i < 72; ++i) P[i] = (const float*)d_in[i];

    const float* s_hat = P[0];
    const float* cqi   = P[1];
    const float* bits  = P[2];
    const float* snr   = P[3];

    char* base = (char*)d_ws;
    size_t off = 0;
    auto alloc = [&](size_t bytes) -> char* {
        off = (off + 255) & ~(size_t)255;
        char* p = base + off;
        off += bytes;
        return p;
    };

    // --- stats + fragment-packed bf16 weights ---
    float* stats = (float*)alloc(256 * sizeof(float));
    zero_f32_kernel<<<1, 256, 0, stream>>>(stats, 256);

    __bf16* w_in2 = pack_weight(stream, P[6], 128, 128, base, off);
    __bf16 *w_m1[3], *w_m2[3], *w_u1[3], *w_u2[3];
    for (int gi = 0; gi < 3; ++gi) {
        int pb = 10 + gi * 10;
        w_m1[gi] = pack_weight(stream, P[pb + 0], 128, 128, base, off);
        w_m2[gi] = pack_weight(stream, P[pb + 2], 128, 128, base, off);
        w_u1[gi] = pack_weight(stream, P[pb + 4], 256, 128, base, off);
        w_u2[gi] = pack_weight(stream, P[pb + 6], 128, 128, base, off);
    }
    __bf16* w_a1 = pack_weight(stream, P[40], 128, 64, base, off);
    __bf16 *w_qkv[2], *w_o[2], *w_f1[2], *w_f2[2];
    for (int t = 0; t < 2; ++t) {
        int pb = 44 + t * 12;
        w_qkv[t] = pack_weight(stream, P[pb + 2],  128, 384, base, off);
        w_o[t]   = pack_weight(stream, P[pb + 4],  128, 128, base, off);
        w_f1[t]  = pack_weight(stream, P[pb + 8],  128, 256, base, off);
        w_f2[t]  = pack_weight(stream, P[pb + 10], 256, 128, base, off);
    }
    __bf16* w_ow1 = pack_weight(stream, P[68], 128, 64, base, off);

    // --- activation buffers (all bf16) ---
    const int N = NROWS, R = UROWS;
    __bf16* H   = (__bf16*)alloc((size_t)N * 128 * 2);
    __bf16* T1  = (__bf16*)alloc((size_t)N * 128 * 2);
    __bf16* T2  = (__bf16*)alloc((size_t)N * 128 * 2);   // also reused as AP1 (N x 64)
    __bf16* MM  = (__bf16*)alloc((size_t)R * 128 * 2);
    __bf16* U   = (__bf16*)alloc((size_t)R * 128 * 2);
    __bf16* Y   = (__bf16*)alloc((size_t)R * 128 * 2);
    __bf16* QKV = (__bf16*)alloc((size_t)R * 384 * 2);
    __bf16* AO  = (__bf16*)alloc((size_t)R * 128 * 2);
    __bf16* F1  = (__bf16*)alloc((size_t)R * 256 * 2);
    __bf16* O1  = (__bf16*)alloc((size_t)R * 64 * 2);

    // --- input MLP ---
    input_mlp1_kernel<<<N / 256, 256, 0, stream>>>(s_hat, cqi, bits, snr,
                                                   P[4], P[5], T1, N);
    launch_gemm(stream, T1, nullptr, w_in2, P[7], nullptr, H, N, 128, 128, 0);

    // --- BatchNorm (batch stats) ---
    bn_stats_kernel<<<1024, 256, 0, stream>>>(H, stats, N);
    {
        long total = (long)N * 128;
        bn_apply_kernel<<<(unsigned)(total / 256), 256, 0, stream>>>(
            H, stats, P[8], P[9], total, 1.0f / (float)N);
    }

    // --- GNN layers ---
    for (int gi = 0; gi < 3; ++gi) {
        int pb = 10 + gi * 10;
        launch_gemm(stream, H,  nullptr, w_m1[gi], P[pb + 1], nullptr, T1, N, 128, 128, 1);
        launch_gemm(stream, T1, nullptr, w_m2[gi], P[pb + 3], nullptr, T2, N, 128, 128, 0);
        mean_l_kernel<<<(512 * 16 * 128) / 256, 256, 0, stream>>>(T2, MM);
        launch_gemm(stream, H,  MM,      w_u1[gi], P[pb + 5], nullptr, T1, N, 256, 128, 2);
        launch_gemm(stream, T1, nullptr, w_u2[gi], P[pb + 7], nullptr, T2, N, 128, 128, 0);
        hipLaunchKernelGGL((ln_rows_kernel<true>), dim3(N / 8), dim3(256), 0, stream,
                           H, T2, P[pb + 8], P[pb + 9], H, N);
    }

    // --- attention pooling over L ---
    launch_gemm(stream, H, nullptr, w_a1, P[41], nullptr, T2, N, 128, 64, 1);
    pool_kernel<<<512 * 16, 64, 0, stream>>>(H, T2, P[42], P[43], U);

    // --- transformer encoder (2 layers, pre-norm) ---
    for (int t = 0; t < 2; ++t) {
        int pb = 44 + t * 12;
        hipLaunchKernelGGL((ln_rows_kernel<false>), dim3(R / 8), dim3(256), 0, stream,
                           U, (const __bf16*)nullptr, P[pb + 0], P[pb + 1], Y, R);
        launch_gemm(stream, Y, nullptr, w_qkv[t], P[pb + 3], nullptr, QKV, R, 128, 384, 0);
        attn16_kernel<<<512 * 4, 32, 0, stream>>>(QKV, AO);
        launch_gemm(stream, AO, nullptr, w_o[t], P[pb + 5], U, U, R, 128, 128, 3);
        hipLaunchKernelGGL((ln_rows_kernel<false>), dim3(R / 8), dim3(256), 0, stream,
                           U, (const __bf16*)nullptr, P[pb + 6], P[pb + 7], Y, R);
        launch_gemm(stream, Y,  nullptr, w_f1[t], P[pb + 9],  nullptr, F1, R, 128, 256, 1);
        launch_gemm(stream, F1, nullptr, w_f2[t], P[pb + 11], U,       U,  R, 256, 128, 3);
    }

    // --- output head ---
    launch_gemm(stream, U, nullptr, w_ow1, P[69], nullptr, O1, R, 128, 64, 1);
    out_head_kernel<<<(R + 255) / 256, 256, 0, stream>>>(O1, P[70], P[71],
                                                         (float*)d_out, R);
}
